// LSTM_60988535603820
// MI455X (gfx1250) — compile-verified, matrix operand order checked by reading, not run on hardware
//
#include <hip/hip_runtime.h>
#include <hip/hip_bf16.h>
#include <cstdint>

// ---------------------------------------------------------------------------
// LSTM (B=64, T=512, D=256, H=512, L=4) + linear head, for MI455X (gfx1250).
//   * bf16 WMMA (v_wmma_f32_16x16x32_bf16) with fp32 accumulation everywhere.
//   * Per layer:   gx = A @ W_ih^T + (b_ih+b_hh)   -- one big WMMA GEMM.
//   * Recurrent scan: 4 batch-groups x 32 WGs; each WG owns 16 hidden units,
//     wave g computes gate g's 16 columns; W_hh slice (64x512 bf16 = 64KB)
//     resident in LDS for all 512 steps. Global h-state double-buffered,
//     per-group atomic barrier each step. K-reduction uses 4 independent
//     WMMA accumulators to break the C-operand RAW chain (scan is latency-
//     critical: ~1 wave/SIMD, nothing else hides XDL latency).
// ---------------------------------------------------------------------------

typedef __bf16 bf16;
typedef __bf16  v16bf __attribute__((ext_vector_type(16)));
typedef float   v8f   __attribute__((ext_vector_type(8)));

#define Bsz 64
#define Tsz 512
#define Dsz 256
#define Hsz 512
#define G4H 2048              // 4*H
#define NGRP 4                // batch groups (16 rows each)
#define NWGH 32               // hidden-slice WGs per group (16 cols each)

// ---- workspace layout (bytes) ----
#define GX_OFF   ((size_t)0)                         // 64*512*2048*4 = 256 MB
#define XBF_OFF  ((size_t)268435456)                 // 64*512*256*2  = 16 MB
#define H0_OFF   ((size_t)285212672)                 // 64*512*512*2  = 32 MB
#define H1_OFF   ((size_t)318767104)                 // 32 MB
#define WIH_OFF  ((size_t)352321536)                 // 2048*512*2 = 2 MB
#define WHH_OFF  ((size_t)354418688)                 // 2 MB
#define BS_OFF   ((size_t)356515840)                 // 2048*4 = 8 KB
#define HST_OFF  ((size_t)356524032)                 // 4 grp * 2 buf * 16*512 bf16 = 128 KB
#define BAR_OFF  ((size_t)356655104)                 // 4 counters
#define WS_NEED  ((size_t)356655168)

static __device__ __forceinline__ v8f wmma_bf16(v16bf a, v16bf b, v8f c) {
    return __builtin_amdgcn_wmma_f32_16x16x32_bf16(
        /*neg_a=*/false, a, /*neg_b=*/false, b,
        /*c_mod=*/(short)0, c, /*reuse_a=*/false, /*reuse_b=*/false);
}

static __device__ __forceinline__ float sigm(float x) {
    return 1.0f / (1.0f + __expf(-x));
}

// ---------------------------------------------------------------------------
// fp32 -> bf16 convert (grid-stride)
// ---------------------------------------------------------------------------
__global__ void k_cvt_bf16(const float* __restrict__ s, bf16* __restrict__ d, size_t n) {
    size_t i  = (size_t)blockIdx.x * blockDim.x + threadIdx.x;
    size_t st = (size_t)gridDim.x * blockDim.x;
    for (; i < n; i += st) d[i] = (bf16)s[i];
}

__global__ void k_bsum(const float* __restrict__ a, const float* __restrict__ b,
                       float* __restrict__ d, int n) {
    int i = blockIdx.x * blockDim.x + threadIdx.x;
    if (i < n) d[i] = a[i] + b[i];
}

__global__ void k_zero(unsigned* __restrict__ p, size_t n) {
    size_t i  = (size_t)blockIdx.x * blockDim.x + threadIdx.x;
    size_t st = (size_t)gridDim.x * blockDim.x;
    for (; i < n; i += st) p[i] = 0u;
}

// ---------------------------------------------------------------------------
// gx GEMM:  out[M=32768, N=2048] = A[M,K] @ W[N,K]^T + bias[N]
// block = 128 thr (4 waves). WG tile 64x128, wave tile 32x64 (2x4 frags).
// A (row-major activations) and W (row-major weights = B^T) fragments load
// with the identical per-lane pattern:
//   lane l (0..15)  -> row base+l,    K half k0..k0+15
//   lane l (16..31) -> row base+l-16, K half k0+16..k0+31
// ---------------------------------------------------------------------------
__global__ void k_gemm_gx(const bf16* __restrict__ A, const bf16* __restrict__ W,
                          const float* __restrict__ bias, float* __restrict__ out,
                          int K, int N) {
    const int lane  = threadIdx.x & 31;
    const int wave  = threadIdx.x >> 5;
    const int mw    = blockIdx.y * 64 + (wave >> 1) * 32;   // wave M base
    const int nw    = blockIdx.x * 128 + (wave & 1) * 64;   // wave N base
    const int lrow  = lane & 15;
    const int khalf = (lane >> 4) * 16;

    v8f acc[2][4];
#pragma unroll
    for (int m = 0; m < 2; ++m)
#pragma unroll
        for (int n = 0; n < 4; ++n) acc[m][n] = (v8f)0.0f;

    for (int k0 = 0; k0 < K; k0 += 32) {
        v16bf a0 = *(const v16bf*)(A + (size_t)(mw + lrow) * K + k0 + khalf);
        v16bf a1 = *(const v16bf*)(A + (size_t)(mw + 16 + lrow) * K + k0 + khalf);
        v16bf b0 = *(const v16bf*)(W + (size_t)(nw + lrow) * K + k0 + khalf);
        v16bf b1 = *(const v16bf*)(W + (size_t)(nw + 16 + lrow) * K + k0 + khalf);
        v16bf b2 = *(const v16bf*)(W + (size_t)(nw + 32 + lrow) * K + k0 + khalf);
        v16bf b3 = *(const v16bf*)(W + (size_t)(nw + 48 + lrow) * K + k0 + khalf);
        acc[0][0] = wmma_bf16(a0, b0, acc[0][0]);
        acc[0][1] = wmma_bf16(a0, b1, acc[0][1]);
        acc[0][2] = wmma_bf16(a0, b2, acc[0][2]);
        acc[0][3] = wmma_bf16(a0, b3, acc[0][3]);
        acc[1][0] = wmma_bf16(a1, b0, acc[1][0]);
        acc[1][1] = wmma_bf16(a1, b1, acc[1][1]);
        acc[1][2] = wmma_bf16(a1, b2, acc[1][2]);
        acc[1][3] = wmma_bf16(a1, b3, acc[1][3]);
    }

    // D layout: VGPR i -> row m0 + i + (lane>=16 ? 8 : 0), col n0 + (lane&15)
    const int rsel = (lane >> 4) << 3;
#pragma unroll
    for (int n = 0; n < 4; ++n) {
        const int col = nw + n * 16 + lrow;
        const float bv = bias[col];
#pragma unroll
        for (int m = 0; m < 2; ++m) {
            const int rb = mw + m * 16 + rsel;
#pragma unroll
            for (int i = 0; i < 8; ++i)
                out[(size_t)(rb + i) * N + col] = acc[m][n][i] + bv;
        }
    }
}

// ---------------------------------------------------------------------------
// Recurrent scan. grid = NGRP*NWGH = 128 WGs, block = 128 (4 waves = 4 gates).
// WG (grp, wgh) owns batch rows [grp*16, grp*16+16) and hidden cols
// [wgh*16, wgh*16+16).  Per step, wave g computes the 16x16 z-tile for gate g
// (z = h @ W_hh^T over K=512, weights in LDS, 4 independent accumulators),
// tiles meet in LDS, gate math updates c (LDS-resident) and h, h goes to the
// double-buffered global state and the layer output; per-group atomic barrier.
// ---------------------------------------------------------------------------
__global__ void k_lstm_scan(const float* __restrict__ gx, const bf16* __restrict__ Whh,
                            bf16* __restrict__ hout, bf16* __restrict__ hstate,
                            unsigned* __restrict__ bars, int T) {
    const int grp  = blockIdx.x >> 5;     // 0..3  batch group
    const int wgh  = blockIdx.x & 31;     // 0..31 hidden slice
    const int tid  = threadIdx.x;
    const int lane = tid & 31;
    const int gate = tid >> 5;            // 0..3
    const int lrow  = lane & 15;
    const int khalf = (lane >> 4) * 16;

    __shared__ __align__(32) bf16 lw[64][Hsz];   // rows: gate*16+n -> Whh row gate*H + wgh*16 + n
    __shared__ float zbuf[4][16][16];
    __shared__ float cst[16][16];

    // stage W_hh slice into LDS (64 rows x 512 bf16 = 64 KB), 8B chunks
    for (int idx = tid; idx < 64 * 128; idx += 128) {
        const int r  = idx >> 7;                 // 0..63
        const int c4 = (idx & 127) * 4;          // bf16 col, step 4
        const int gr = (r >> 4) * Hsz + wgh * 16 + (r & 15);
        *(uint2*)&lw[r][c4] = *(const uint2*)(Whh + (size_t)gr * Hsz + c4);
    }
    for (int idx = tid; idx < 256; idx += 128) cst[idx >> 4][idx & 15] = 0.0f;
    __syncthreads();

    unsigned* bar = bars + grp;
    bf16* hbase = hstate + (size_t)grp * (2 * 16 * Hsz);
    const bf16* arow = nullptr;  // per-lane A row base set below
    const size_t aoff = (size_t)lrow * Hsz + khalf;

    for (int t = 0; t < T; ++t) {
        const bf16* hin = hbase + (size_t)(t & 1) * (16 * Hsz);        // written at t-1
        bf16*       hwr = hbase + (size_t)((t & 1) ^ 1) * (16 * Hsz);  // for step t
        arow = hin + aoff;
        const bf16* wrow = &lw[gate * 16 + lrow][khalf];

        // 4 independent accumulator chains (k-iterations striped mod 4) so the
        // scheduler can interleave WMMAs instead of one 16-deep RAW chain.
        v8f acc0 = (v8f)0.0f, acc1 = (v8f)0.0f, acc2 = (v8f)0.0f, acc3 = (v8f)0.0f;
#pragma unroll
        for (int kb = 0; kb < Hsz; kb += 128) {
            v16bf a0 = *(const v16bf*)(arow + kb);
            v16bf a1 = *(const v16bf*)(arow + kb + 32);
            v16bf a2 = *(const v16bf*)(arow + kb + 64);
            v16bf a3 = *(const v16bf*)(arow + kb + 96);
            v16bf b0 = *(const v16bf*)(wrow + kb);
            v16bf b1 = *(const v16bf*)(wrow + kb + 32);
            v16bf b2 = *(const v16bf*)(wrow + kb + 64);
            v16bf b3 = *(const v16bf*)(wrow + kb + 96);
            acc0 = wmma_bf16(a0, b0, acc0);
            acc1 = wmma_bf16(a1, b1, acc1);
            acc2 = wmma_bf16(a2, b2, acc2);
            acc3 = wmma_bf16(a3, b3, acc3);
        }
        v8f acc = (acc0 + acc1) + (acc2 + acc3);

        {
            const int rb = (lane >> 4) << 3;
#pragma unroll
            for (int i = 0; i < 8; ++i) zbuf[gate][rb + i][lrow] = acc[i];
        }
        __syncthreads();

        // gate stage: 16x16 = 256 elements, 2 per thread (fixed mapping -> c stays thread-private)
        for (int e = tid; e < 256; e += 128) {
            const int bl = e >> 4, j = e & 15;
            const size_t gxo =
                ((size_t)(grp * 16 + bl) * Tsz + t) * G4H + (size_t)wgh * 16 + j;
            const float g0 = gx[gxo];
            const float g1 = gx[gxo + Hsz];
            const float g2 = gx[gxo + 2 * Hsz];
            const float g3 = gx[gxo + 3 * Hsz];
            const float zi = zbuf[0][bl][j] + g0;
            const float zf = zbuf[1][bl][j] + g1;
            const float zg = zbuf[2][bl][j] + g2;
            const float zo = zbuf[3][bl][j] + g3;
            const float ii = sigm(zi);
            const float ff = sigm(zf);
            const float gg = tanhf(zg);
            const float oo = sigm(zo);
            const float c = ff * cst[bl][j] + ii * gg;
            cst[bl][j] = c;
            const float h = oo * tanhf(c);
            const bf16 hb = (bf16)h;
            hwr[(size_t)bl * Hsz + wgh * 16 + j] = hb;
            hout[((size_t)(grp * 16 + bl) * Tsz + t) * Hsz + wgh * 16 + j] = hb;
        }
        __syncthreads();

        // per-group global barrier (32 resident WGs)
        if (tid == 0) {
            __threadfence();
            atomicAdd(bar, 1u);
            const unsigned target = (unsigned)(t + 1) * NWGH;
            while (*((volatile unsigned*)bar) < target) __builtin_amdgcn_s_sleep(1);
        }
        __syncthreads();
    }
}

// ---------------------------------------------------------------------------
// Linear head on last timestep: out[b] = h[b,T-1,:] . Wout + bout   (O=1)
// ---------------------------------------------------------------------------
__global__ void k_head(const bf16* __restrict__ h, const float* __restrict__ Wout,
                       const float* __restrict__ bout, float* __restrict__ out) {
    const int b = threadIdx.x;   // 64 threads
    if (b >= Bsz) return;
    const bf16* hp = h + ((size_t)b * Tsz + (Tsz - 1)) * Hsz;
    float s = 0.0f;
    for (int j = 0; j < Hsz; ++j) s += (float)hp[j] * Wout[j];
    out[b] = s + bout[0];
}

// ---------------------------------------------------------------------------
extern "C" void kernel_launch(void* const* d_in, const int* in_sizes, int n_in,
                              void* d_out, int out_size, void* d_ws, size_t ws_size,
                              hipStream_t stream) {
    (void)in_sizes; (void)n_in; (void)out_size;
    if (ws_size < WS_NEED) return;   // insufficient scratch; avoid OOB

    const float* x = (const float*)d_in[0];
    const float *Wih[4], *Whh[4], *bih[4], *bhh[4];
    for (int l = 0; l < 4; ++l) {
        Wih[l] = (const float*)d_in[1 + 4 * l];
        Whh[l] = (const float*)d_in[2 + 4 * l];
        bih[l] = (const float*)d_in[3 + 4 * l];
        bhh[l] = (const float*)d_in[4 + 4 * l];
    }
    const float* Wout = (const float*)d_in[17];
    const float* bout = (const float*)d_in[18];

    char* ws = (char*)d_ws;
    float*    gxb   = (float*)(ws + GX_OFF);
    bf16*     xbf   = (bf16*)(ws + XBF_OFF);
    bf16*     hbuf0 = (bf16*)(ws + H0_OFF);
    bf16*     hbuf1 = (bf16*)(ws + H1_OFF);
    bf16*     wihb  = (bf16*)(ws + WIH_OFF);
    bf16*     whhb  = (bf16*)(ws + WHH_OFF);
    float*    bsum  = (float*)(ws + BS_OFF);
    bf16*     hst   = (bf16*)(ws + HST_OFF);
    unsigned* bars  = (unsigned*)(ws + BAR_OFF);

    k_cvt_bf16<<<256, 256, 0, stream>>>(x, xbf, (size_t)Bsz * Tsz * Dsz);

    for (int l = 0; l < 4; ++l) {
        const int K = (l == 0) ? Dsz : Hsz;
        k_cvt_bf16<<<256, 256, 0, stream>>>(Wih[l], wihb, (size_t)G4H * K);
        k_cvt_bf16<<<256, 256, 0, stream>>>(Whh[l], whhb, (size_t)G4H * Hsz);
        k_bsum<<<8, 256, 0, stream>>>(bih[l], bhh[l], bsum, G4H);

        const bf16* Ain = (l == 0) ? xbf : ((l & 1) ? hbuf0 : hbuf1);
        bf16* Hout = (l & 1) ? hbuf1 : hbuf0;

        // gx = Ain @ Wih^T + bsum : M=32768, N=2048
        k_gemm_gx<<<dim3(G4H / 128, (Bsz * Tsz) / 64), 128, 0, stream>>>(
            Ain, wihb, bsum, gxb, K, G4H);

        // reset h-state (both buffers) + barrier counters (contiguous in ws)
        k_zero<<<64, 256, 0, stream>>>((unsigned*)hst, (size_t)(131072 + 16) / 4);

        k_lstm_scan<<<NGRP * NWGH, 128, 0, stream>>>(gxb, whhb, Hout, hst, bars, Tsz);
    }

    k_head<<<1, 64, 0, stream>>>(hbuf1, Wout, bout, (float*)d_out);
}
